// PNN_43095701848191
// MI455X (gfx1250) — matrix operand order repaired and checked by reading, not compiled, not run on hardware
//
#include <hip/hip_runtime.h>

// ---------------------------------------------------------------------------
// PNN on MI455X (gfx1250): fused  normalize -> bf16 WMMA GEMM -> Gauss ->
// per-class sums -> row-normalize/argmax.  G matrix (128MB) never touches HBM.
//
// GEMM structure: per workgroup (8 wave32), 128 test rows x 128 train rows
// n-blocks; LDS holds a 128x64 bf16 train tile; each A-fragment is reused by
// 8 WMMAs (8 n-tiles), so operand-prep VALU is amortized 8x.
// If ws_size permits, operands are pre-normalized/converted to bf16 once and
// the LDS tile is staged with global_load_async_to_lds_b128 (ASYNCcnt path).
// ---------------------------------------------------------------------------

typedef __attribute__((ext_vector_type(16))) __bf16 v16bf;
typedef __attribute__((ext_vector_type(8)))  __bf16 v8bf;
typedef __attribute__((ext_vector_type(8)))  float  v8f;

#if __has_builtin(__builtin_amdgcn_global_load_async_to_lds_b128)
#define PNN_HAVE_ASYNC_LDS 1
// builtin signature (from hipcc diagnostic): first param is
//   int __attribute__((vector_size(16))) addrspace(1)*  (global src)
// second is the LDS destination in addrspace(3).
typedef int pnn_v4i __attribute__((vector_size(16)));
typedef __attribute__((address_space(1))) pnn_v4i* pnn_gv4i_p;
typedef __attribute__((address_space(3))) pnn_v4i* pnn_lv4i_p;
#endif

#define NTRAIN 8192
#define NTEST  4096
#define KDIM   1024
#define NB_ROWS 128                 // train rows per n-block (8 WMMA n-tiles)
#define KSLICE  64                  // K elements staged per LDS tile
#define LDSTRIDE 72                 // bf16 row stride in LDS (64 + 8 pad = 144B)

// ---------------------------------------------------------------------------
// Kernel 0: zero the prob accumulator (8192 floats)
// ---------------------------------------------------------------------------
__global__ void pnn_init_kernel(float* __restrict__ prob) {
    int i = blockIdx.x * blockDim.x + threadIdx.x;
    if (i < NTEST * 2) prob[i] = 0.0f;
}

// ---------------------------------------------------------------------------
// Kernel 1a (fallback): inverse L2 norms (train rows 0..8191, test 8192..12287)
// ---------------------------------------------------------------------------
__global__ void pnn_norm_kernel(const float* __restrict__ tr,
                                const float* __restrict__ te,
                                float* __restrict__ winv) {
    int row = blockIdx.x;
    const float* src = (row < NTRAIN) ? tr + (size_t)row * KDIM
                                      : te + (size_t)(row - NTRAIN) * KDIM;
    int t = threadIdx.x;
    float s = 0.0f;
#pragma unroll
    for (int i = 0; i < 4; ++i) {
        float x = src[t + i * 256];
        s += x * x;
    }
#pragma unroll
    for (int off = 16; off > 0; off >>= 1) s += __shfl_down(s, off, 32);
    __shared__ float ps[8];
    if ((t & 31) == 0) ps[t >> 5] = s;
    __syncthreads();
    if (t == 0) {
        float tot = 0.0f;
#pragma unroll
        for (int i = 0; i < 8; ++i) tot += ps[i];
        winv[row] = rsqrtf(tot);
    }
}

// ---------------------------------------------------------------------------
// Kernel 1b (pre path): L2-normalize every row and write bf16 copies.
// ---------------------------------------------------------------------------
__global__ void pnn_cvt_kernel(const float* __restrict__ tr,
                               const float* __restrict__ te,
                               __bf16* __restrict__ trBf,
                               __bf16* __restrict__ teBf) {
    int row = blockIdx.x;
    const float* src;
    __bf16* dst;
    if (row < NTRAIN) { src = tr + (size_t)row * KDIM; dst = trBf + (size_t)row * KDIM; }
    else { src = te + (size_t)(row - NTRAIN) * KDIM; dst = teBf + (size_t)(row - NTRAIN) * KDIM; }

    int t = threadIdx.x;
    float x[4];
    float s = 0.0f;
#pragma unroll
    for (int i = 0; i < 4; ++i) {
        x[i] = src[t + i * 256];
        s += x[i] * x[i];
    }
#pragma unroll
    for (int off = 16; off > 0; off >>= 1) s += __shfl_down(s, off, 32);
    __shared__ float ps[8];
    if ((t & 31) == 0) ps[t >> 5] = s;
    __syncthreads();
    float tot = 0.0f;
#pragma unroll
    for (int i = 0; i < 8; ++i) tot += ps[i];
    float inv = rsqrtf(tot);
#pragma unroll
    for (int i = 0; i < 4; ++i) dst[t + i * 256] = (__bf16)(x[i] * inv);
}

// ---------------------------------------------------------------------------
// Fragment helpers (per-lane ISA layouts, 16-bit operands)
// A 16x32: lane = M row; lanes<16 hold K {0..7,16..23}, lanes>=16 {8..15,24..31}
// B 32x16: lane = N col; lanes<16 hold K {0..7,16..23}, lanes>=16 {8..15,24..31}
// ---------------------------------------------------------------------------
__device__ inline v16bf pnn_join(v8bf lo, v8bf hi) {
    v16bf r;
#pragma unroll
    for (int i = 0; i < 8; ++i) { r[i] = lo[i]; r[8 + i] = hi[i]; }
    return r;
}

__device__ inline v16bf load_a_pre(const __bf16* __restrict__ p) {
    return pnn_join(*(const v8bf*)p, *(const v8bf*)(p + 16));
}

__device__ inline v16bf load_a_cvt(const float* __restrict__ p, float s) {
    const float4* pa = (const float4*)p;
    float4 f0 = pa[0], f1 = pa[1];
    const float4* pb = (const float4*)(p + 16);
    float4 f2 = pb[0], f3 = pb[1];
    v16bf a;
    a[0]  = (__bf16)(f0.x * s); a[1]  = (__bf16)(f0.y * s);
    a[2]  = (__bf16)(f0.z * s); a[3]  = (__bf16)(f0.w * s);
    a[4]  = (__bf16)(f1.x * s); a[5]  = (__bf16)(f1.y * s);
    a[6]  = (__bf16)(f1.z * s); a[7]  = (__bf16)(f1.w * s);
    a[8]  = (__bf16)(f2.x * s); a[9]  = (__bf16)(f2.y * s);
    a[10] = (__bf16)(f2.z * s); a[11] = (__bf16)(f2.w * s);
    a[12] = (__bf16)(f3.x * s); a[13] = (__bf16)(f3.y * s);
    a[14] = (__bf16)(f3.z * s); a[15] = (__bf16)(f3.w * s);
    return a;
}

__device__ inline v16bf load_b_frag(const __bf16* __restrict__ p) {
    return pnn_join(*(const v8bf*)p, *(const v8bf*)(p + 16));
}

// ---------------------------------------------------------------------------
// Kernel 2: fused GEMM (bf16 WMMA) + Gauss + per-class accumulation.
// grid = (NTEST/128, NTRAIN/1024), block = 256 (8 wave32s).
// ---------------------------------------------------------------------------
template <bool PRE>
__global__ void __launch_bounds__(256)
pnn_gemm_kernel(const float*  __restrict__ te,
                const float*  __restrict__ tr,
                const __bf16* __restrict__ teBf,
                const __bf16* __restrict__ trBf,
                const int*    __restrict__ labels,
                const float*  __restrict__ winv,
                float*        __restrict__ prob) {
    __shared__ __bf16 ldsB[NB_ROWS * LDSTRIDE];   // 18 KB train tile

    const int tid  = threadIdx.x;
    const int lane = tid & 31;
    const int wave = tid >> 5;
    const int half = lane >> 4;                   // 0 | 1
    const int lr   = lane & 15;

    const int chunkBase = blockIdx.y * 1024;      // 1024 train rows per chunk
    const int tb        = blockIdx.x * 128 + wave * 16;

    // per-lane fixed A row (test row)
    const int rowA = tb + lr;
    const int kA   = half * 8;                    // A K-offset within 32-slice
    const int kB   = half * 8;                    // B K-offset within 32-slice

    const float*  teRow   = te   ? te   + (size_t)rowA * KDIM : nullptr;
    const __bf16* teBfRow = teBf ? teBf + (size_t)rowA * KDIM : nullptr;
    const float   sA      = winv ? winv[NTRAIN + rowA] : 0.0f;

    float acc0[8], acc1[8];
#pragma unroll
    for (int r = 0; r < 8; ++r) { acc0[r] = 0.0f; acc1[r] = 0.0f; }

    for (int nb = 0; nb < 1024 / NB_ROWS; ++nb) {
        const int nbase = chunkBase + nb * NB_ROWS;

        v8f acc[8];
#pragma unroll
        for (int nt = 0; nt < 8; ++nt) acc[nt] = {};

        for (int ks = 0; ks < KDIM / KSLICE; ++ks) {
            const int ksb = ks * KSLICE;
            __syncthreads();                      // previous tile fully consumed

            // ---- stage 128 x 64 bf16 train tile into LDS
#pragma unroll
            for (int i = 0; i < 4; ++i) {
                int c   = tid + i * 256;          // 1024 chunks of 8 elems
                int row = c >> 3;
                int col = (c & 7) << 3;
                __bf16* dst = &ldsB[row * LDSTRIDE + col];
                if constexpr (PRE) {
#ifdef PNN_HAVE_ASYNC_LDS
                    const __bf16* src = trBf + (size_t)(nbase + row) * KDIM + ksb + col;
                    __builtin_amdgcn_global_load_async_to_lds_b128(
                        (pnn_gv4i_p)src, (pnn_lv4i_p)dst, 0, 0);
#else
                    const v8bf* src =
                        (const v8bf*)(trBf + (size_t)(nbase + row) * KDIM + ksb + col);
                    *(v8bf*)dst = *src;
#endif
                } else {
                    int gr = nbase + row;
                    const float* src = tr + (size_t)gr * KDIM + ksb + col;
                    float s = winv[gr];
                    float4 f0 = ((const float4*)src)[0];
                    float4 f1 = ((const float4*)src)[1];
                    v8bf v;
                    v[0] = (__bf16)(f0.x * s); v[1] = (__bf16)(f0.y * s);
                    v[2] = (__bf16)(f0.z * s); v[3] = (__bf16)(f0.w * s);
                    v[4] = (__bf16)(f1.x * s); v[5] = (__bf16)(f1.y * s);
                    v[6] = (__bf16)(f1.z * s); v[7] = (__bf16)(f1.w * s);
                    *(v8bf*)dst = v;
                }
            }
            // prefetch next K-slice of this n-block (global_prefetch_b8)
            if (ks + 1 < KDIM / KSLICE) {
                int prow = nbase + (tid & 127);
                int pcol = ksb + KSLICE + ((tid >> 7) << 5);
                if constexpr (PRE)
                    __builtin_prefetch(trBf + (size_t)prow * KDIM + pcol, 0, 1);
                else
                    __builtin_prefetch(tr + (size_t)prow * KDIM + pcol, 0, 1);
            }
#if defined(PNN_HAVE_ASYNC_LDS)
            if constexpr (PRE) {
#if __has_builtin(__builtin_amdgcn_s_wait_asynccnt)
                __builtin_amdgcn_s_wait_asynccnt(0);
#else
                asm volatile("s_wait_asynccnt 0x0" ::: "memory");
#endif
            }
#endif
            __syncthreads();                      // tile visible to all waves

            // ---- 2 K-steps x 8 n-tiles: one A-frag feeds 8 WMMAs
#pragma unroll
            for (int kt = 0; kt < 2; ++kt) {
                const int k0 = ksb + kt * 32;
                v16bf a;
                if constexpr (PRE) a = load_a_pre(teBfRow + k0 + kA);
                else               a = load_a_cvt(teRow + k0 + kA, sA);
#pragma unroll
                for (int nt = 0; nt < 8; ++nt) {
                    v16bf b = load_b_frag(&ldsB[(nt * 16 + lr) * LDSTRIDE + kt * 32 + kB]);
                    acc[nt] = __builtin_amdgcn_wmma_f32_16x16x32_bf16(
                                  false, a, false, b, (short)0, acc[nt], false, false);
                }
            }
        }

        // ---- epilogue: Gauss + class accumulation (lane's train col = lr)
#pragma unroll
        for (int nt = 0; nt < 8; ++nt) {
            const float lbl = (float)labels[nbase + nt * 16 + lr];
            const float nlb = 1.0f - lbl;
#pragma unroll
            for (int r = 0; r < 8; ++r) {
                float d2 = fmaxf(2.0f - 2.0f * acc[nt][r], 0.0f);
                float g  = __expf(-2.0f * sqrtf(d2));  // exp(-D/(2*s^2)), s=0.5
                acc1[r] += g * lbl;
                acc0[r] += g * nlb;
            }
        }
    }

    // ---- cross-lane reduce over the 16 train-column lanes, then atomics
#pragma unroll
    for (int r = 0; r < 8; ++r) {
        float a0 = acc0[r], a1 = acc1[r];
#pragma unroll
        for (int off = 8; off > 0; off >>= 1) {
            a0 += __shfl_xor(a0, off, 16);
            a1 += __shfl_xor(a1, off, 16);
        }
        if (lr == 0) {
            int row = tb + half * 8 + r;          // C-layout: VGPR r -> M = r (+8)
            atomicAdd(&prob[row * 2 + 0], a0);
            atomicAdd(&prob[row * 2 + 1], a1);
        }
    }
}

// ---------------------------------------------------------------------------
// Kernel 3: row-normalize prob, argmax -> d_out = [results(4096) | prob(4096x2)]
// ---------------------------------------------------------------------------
__global__ void pnn_finalize_kernel(const float* __restrict__ prob,
                                    float* __restrict__ out) {
    int p = blockIdx.x * blockDim.x + threadIdx.x;
    if (p < NTEST) {
        float p0 = prob[p * 2 + 0];
        float p1 = prob[p * 2 + 1];
        float s  = p0 + p1;
        out[p] = (p1 > p0) ? 1.0f : 0.0f;
        out[NTEST + p * 2 + 0] = p0 / s;
        out[NTEST + p * 2 + 1] = p1 / s;
    }
}

// ---------------------------------------------------------------------------
extern "C" void kernel_launch(void* const* d_in, const int* in_sizes, int n_in,
                              void* d_out, int out_size, void* d_ws, size_t ws_size,
                              hipStream_t stream) {
    const float* tr  = (const float*)d_in[0];
    const int*   lbl = (const int*)d_in[1];
    const float* te  = (const float*)d_in[2];
    float* out = (float*)d_out;

    const dim3 ggrid(NTEST / 128, NTRAIN / 1024);
    const size_t wsPre = (size_t)(NTRAIN + NTEST) * KDIM * sizeof(__bf16)
                       + (size_t)NTEST * 2 * sizeof(float);

    if (ws_size >= wsPre) {
        // pre-converted bf16 path (async-LDS staging, zero in-loop conversion)
        __bf16* trBf = (__bf16*)d_ws;
        __bf16* teBf = trBf + (size_t)NTRAIN * KDIM;
        float*  prob = (float*)(teBf + (size_t)NTEST * KDIM);
        pnn_init_kernel<<<(NTEST * 2 + 255) / 256, 256, 0, stream>>>(prob);
        pnn_cvt_kernel<<<NTRAIN + NTEST, 256, 0, stream>>>(tr, te, trBf, teBf);
        pnn_gemm_kernel<true><<<ggrid, 256, 0, stream>>>(
            nullptr, nullptr, teBf, trBf, lbl, nullptr, prob);
        pnn_finalize_kernel<<<(NTEST + 255) / 256, 256, 0, stream>>>(prob, out);
    } else {
        // small-workspace fallback: on-the-fly f32 -> bf16 conversion
        float* W    = (float*)d_ws;        // [0,12288): inv L2 norms (train|test)
        float* prob = W + 12288;           // [12288,20480): class accumulators
        pnn_init_kernel<<<(NTEST * 2 + 255) / 256, 256, 0, stream>>>(prob);
        pnn_norm_kernel<<<NTRAIN + NTEST, 256, 0, stream>>>(tr, te, W);
        pnn_gemm_kernel<false><<<ggrid, 256, 0, stream>>>(
            te, tr, nullptr, nullptr, lbl, W, prob);
        pnn_finalize_kernel<<<(NTEST + 255) / 256, 256, 0, stream>>>(prob, out);
    }
}
